// VectorQuantizerEMA_66571993088828
// MI455X (gfx1250) — matrix operand order repaired and checked by reading, not compiled, not run on hardware
//
#include <hip/hip_runtime.h>
#include <hip/hip_bf16.h>

// ---------------- problem constants (from reference) ----------------
#define K_CODES   512
#define EMB       64
#define T_LEN     4096
#define B_BATCH   32
#define N_ROWS    (B_BATCH * T_LEN)      // 131072
#define LOSS_PARTS (N_ROWS / 256)        // 512 blocks in pass B
#define DECAY_F   0.99f
#define ONE_M_DECAY 0.01f
#define EPSILON_F 1e-05f
#define COMMIT_F  0.25f
#define D_BIAS    1024.0f                // makes all packed distances positive

typedef __attribute__((ext_vector_type(16))) __bf16 v16bf;
typedef __attribute__((ext_vector_type(8)))  __bf16 v8bf;
typedef __attribute__((ext_vector_type(8)))  float  v8f;

// f32 -> bf16 round-to-nearest-even (storage only, no __bf16 arithmetic)
static __device__ __forceinline__ __bf16 f2bf(float f) {
    unsigned u = __builtin_bit_cast(unsigned, f);
    u += 0x7FFFu + ((u >> 16) & 1u);
    unsigned short h = (unsigned short)(u >> 16);
    __bf16 out;
    __builtin_memcpy(&out, &h, 2);
    return out;
}

// ---------------- pass A: ||e||^2 per code (f32 exact) ----------------
__global__ __launch_bounds__(256) void vq_e2_kernel(const float* __restrict__ emb,
                                                    float* __restrict__ e2g) {
    int k = blockIdx.x * 256 + threadIdx.x;
    if (k < K_CODES) {
        const float* row = emb + (size_t)k * EMB;
        float s = 0.f;
        #pragma unroll
        for (int c = 0; c < EMB; ++c) { float v = row[c]; s += v * v; }
        e2g[k] = s;
    }
}

// ---------------- pass B: WMMA distance search + quantize + loss ----------------
// 8 waves/block, 32 z-rows per wave (two B-fragment sets), grid = N_ROWS/256 = 512.
// A (16x32 bf16) = codebook tile (codes along M), B (32x16 bf16) = z tiles (rows along N).
// D(16x16 f32): lane owns one z-row across 8 VGPRs. Argmin via monotonic unsigned
// key packing: d+1024 > 0, so float ordering == unsigned ordering; low 9 bits carry
// the code id so v_min_u32 does compare+select+tie-break(lowest index) in one op.
__global__ __launch_bounds__(256) void vq_assign_kernel(
    const float* __restrict__ z, const float* __restrict__ emb,
    const float* __restrict__ e2g,
    float* __restrict__ out_q, float* __restrict__ out_idx_f,
    int* __restrict__ ws_idx, float* __restrict__ ws_loss) {

    __shared__ __align__(16) __bf16 semb[K_CODES * EMB];  // 64 KB bf16 codebook
    __shared__ __align__(16) float se2b[K_CODES];          // ||e||^2 + D_BIAS
    __shared__ float swred[8];

    const int tid = threadIdx.x;
    for (int i = tid; i < K_CODES * EMB; i += 256) semb[i] = f2bf(emb[i]);
    for (int i = tid; i < K_CODES; i += 256) se2b[i] = e2g[i] + D_BIAS;
    __syncthreads();

    const int wave = tid >> 5;
    const int lane = tid & 31;
    const int lrow = lane & 15;
    const int chi  = lane >> 4;                   // half-wave selector
    const int base = blockIdx.x * 256 + wave * 32;
    const int n0   = base + lrow;                 // row set 0
    const int n1   = base + 16 + lrow;            // row set 1
    const int b0 = n0 >> 12, t0 = n0 & (T_LEN - 1);
    const int b1 = n1 >> 12, t1 = n1 & (T_LEN - 1);
    const float* zr0 = z + (size_t)b0 * EMB * T_LEN + t0;  // stride T_LEN per channel
    const float* zr1 = z + (size_t)b1 * EMB * T_LEN + t1;

    // B fragments (loop-invariant): lanes 0-15 -> K {0..7,16..23}, lanes 16-31 -> K {8..15,24..31}
    v16bf b0a, b1a, b0b, b1b;
    #pragma unroll
    for (int e = 0; e < 16; ++e) {
        int c0 = chi * 8 + e + (e >= 8 ? 8 : 0);
        b0a[e] = f2bf(zr0[(size_t)c0 * T_LEN]);
        b1a[e] = f2bf(zr0[(size_t)(c0 + 32) * T_LEN]);
        b0b[e] = f2bf(zr1[(size_t)c0 * T_LEN]);
        b1b[e] = f2bf(zr1[(size_t)(c0 + 32) * T_LEN]);
    }

    unsigned key0 = 0xFFFFFFFFu, key1 = 0xFFFFFFFFu;
    for (int kt = 0; kt < K_CODES / 16; ++kt) {
        // A layout: lane<16: M=lrow, K={0..7,16..23}; lane>=16: M=lrow, K={8..15,24..31}
        const __bf16* ap = &semb[(kt * 16 + lrow) * EMB + chi * 8];
        v8bf a00 = *(const v8bf*)(ap);        // C chunk0
        v8bf a01 = *(const v8bf*)(ap + 16);
        v8bf a10 = *(const v8bf*)(ap + 32);   // C chunk1 (cc=32)
        v8bf a11 = *(const v8bf*)(ap + 48);
        v16bf A0, A1;
        #pragma unroll
        for (int i = 0; i < 8; ++i) {
            A0[i] = a00[i]; A0[i + 8] = a01[i];
            A1[i] = a10[i]; A1[i + 8] = a11[i];
        }
        // two independent WMMA chains (row set 0 / row set 1) per A fetch
        v8f acc0 = {}, acc1 = {};
        acc0 = __builtin_amdgcn_wmma_f32_16x16x32_bf16(false, A0, false, b0a,
                                                       (short)0, acc0, false, false);
        acc1 = __builtin_amdgcn_wmma_f32_16x16x32_bf16(false, A0, false, b0b,
                                                       (short)0, acc1, false, false);
        acc0 = __builtin_amdgcn_wmma_f32_16x16x32_bf16(false, A1, false, b1a,
                                                       (short)0, acc0, false, false);
        acc1 = __builtin_amdgcn_wmma_f32_16x16x32_bf16(false, A1, false, b1b,
                                                       (short)0, acc1, false, false);

        const int sb = kt * 16 + chi * 8;     // 8 contiguous biased ||e||^2 values
        const float4 s0 = *(const float4*)&se2b[sb];
        const float4 s1 = *(const float4*)&se2b[sb + 4];
        const float sc[8] = {s0.x, s0.y, s0.z, s0.w, s1.x, s1.y, s1.z, s1.w};
        const unsigned cb = (unsigned)sb;     // codebase; cb|r == cb+r (cb mult of 8)
        #pragma unroll
        for (int r = 0; r < 8; ++r) {
            float d0 = fmaf(-2.0f, acc0[r], sc[r]);      // biased positive distance
            float d1 = fmaf(-2.0f, acc1[r], sc[r]);
            unsigned k0 = (__builtin_bit_cast(unsigned, d0) & 0xFFFFFE00u) | (cb | (unsigned)r);
            unsigned k1 = (__builtin_bit_cast(unsigned, d1) & 0xFFFFFE00u) | (cb | (unsigned)r);
            key0 = min(key0, k0);
            key1 = min(key1, k1);
        }
    }
    // combine the two lanes holding the same z-row (unsigned min keeps lowest index on tie)
    key0 = min(key0, (unsigned)__shfl_xor((int)key0, 16, 32));
    key1 = min(key1, (unsigned)__shfl_xor((int)key1, 16, 32));
    const int mini0 = (int)(key0 & 511u);
    const int mini1 = (int)(key1 & 511u);

    // lane L finalizes row base+L (L<16 -> set0 row, L>=16 -> set1 row base+16+(L-16))
    const int nq = base + lane;
    const int bq = nq >> 12, tq = nq & (T_LEN - 1);
    const int qmini = (lane < 16) ? mini0 : mini1;
    out_idx_f[nq] = (float)qmini;   // indices output (as float per d_out dtype)
    ws_idx[nq]    = qmini;          // int copy for the segment-sum pass

    // quantized output (exact f32 gather) + commitment loss partial
    const float* erow = emb + (size_t)qmini * EMB;
    const float* zq = z + (size_t)bq * EMB * T_LEN + tq;
    float* qq = out_q + (size_t)bq * EMB * T_LEN + tq;
    float lsum = 0.f;
    #pragma unroll
    for (int c = 0; c < EMB; ++c) {
        float qv = erow[c];
        float zv = zq[(size_t)c * T_LEN];
        qq[(size_t)c * T_LEN] = qv;              // straight-through fwd value == quantized
        float dd = zv - qv;
        lsum += dd * dd;
    }
    // fixed-order block reduction -> per-block partial (deterministic, no fp atomics)
    #pragma unroll
    for (int off = 16; off > 0; off >>= 1) lsum += __shfl_xor(lsum, off, 32);
    if (lane == 0) swred[wave] = lsum;
    __syncthreads();
    if (tid == 0) {
        float s = 0.f;
        #pragma unroll
        for (int w = 0; w < 8; ++w) s += swred[w];
        ws_loss[blockIdx.x] = s;
    }
}

// ---------------- pass C: deterministic segment-sum (counts + embed_sum) ----------------
// One block per code; index array (512 KB) stays hot in the 192 MB L2 across 512 blocks.
__global__ __launch_bounds__(512) void vq_segsum_kernel(
    const int* __restrict__ idx, const float* __restrict__ z,
    float* __restrict__ cnts, float* __restrict__ esum) {

    const int k   = blockIdx.x;
    const int tid = threadIdx.x;
    const int c   = tid & 63;      // embedding component
    const int sub = tid >> 6;      // 8 sub-groups of 64 threads
    __shared__ int   sidx[512];
    __shared__ float part[8][64];
    __shared__ float pcnt[8];

    float acc = 0.f, cnt = 0.f;
    for (int bse = 0; bse < N_ROWS; bse += 512) {
        sidx[tid] = idx[bse + tid];
        __syncthreads();
        const int j0 = sub * 64;
        #pragma unroll 4
        for (int j = j0; j < j0 + 64; ++j) {
            if (sidx[j] == k) {
                int n = bse + j;
                int b = n >> 12, t = n & (T_LEN - 1);
                acc += z[(size_t)b * EMB * T_LEN + (size_t)c * T_LEN + t];
                if (c == 0) cnt += 1.f;
            }
        }
        __syncthreads();
    }
    part[sub][c] = acc;
    if (c == 0) pcnt[sub] = cnt;
    __syncthreads();
    if (sub == 0) {
        float s = 0.f;
        #pragma unroll
        for (int q = 0; q < 8; ++q) s += part[q][c];   // fixed order
        esum[(size_t)k * EMB + c] = s;
        if (c == 0) {
            float sc = 0.f;
            #pragma unroll
            for (int q = 0; q < 8; ++q) sc += pcnt[q];
            cnts[k] = sc;
        }
    }
}

// ---------------- pass D: EMA update, perplexity, loss (1 block) ----------------
__global__ __launch_bounds__(256) void vq_final_kernel(
    const float* __restrict__ cluster_size, const float* __restrict__ embed_avg,
    const float* __restrict__ cnts, const float* __restrict__ esum,
    const float* __restrict__ loss_part,
    float* __restrict__ out_loss, float* __restrict__ out_perp,
    float* __restrict__ out_newemb, float* __restrict__ out_newcs,
    float* __restrict__ out_newavg) {

    __shared__ float sn[K_CODES];
    __shared__ float scnt[K_CODES];
    __shared__ float s_ntot;
    const int tid = threadIdx.x;

    for (int k = tid; k < K_CODES; k += 256) {
        float nk  = cnts[k];
        float ncs = cluster_size[k] * DECAY_F + ONE_M_DECAY * nk;
        scnt[k] = nk;
        sn[k]   = ncs;
        out_newcs[k] = ncs;
    }
    __syncthreads();
    if (tid == 0) {
        float ntot = 0.f;
        for (int k = 0; k < K_CODES; ++k) ntot += sn[k];   // fixed order
        s_ntot = ntot;
        float plog = 0.f;
        for (int k = 0; k < K_CODES; ++k) {
            float p = scnt[k] / (float)N_ROWS;
            plog += p * logf(p + 1e-10f);
        }
        out_perp[0] = expf(-plog);
        float ls = 0.f;
        for (int i = 0; i < LOSS_PARTS; ++i) ls += loss_part[i];
        out_loss[0] = COMMIT_F * ls / (float)((size_t)B_BATCH * EMB * T_LEN);
    }
    __syncthreads();
    const float ntot = s_ntot;
    const float denom = ntot + (float)K_CODES * EPSILON_F;
    for (int i = tid; i < K_CODES * EMB; i += 256) {
        int k = i >> 6;
        float cs = (sn[k] + EPSILON_F) / denom * ntot;
        float na = embed_avg[i] * DECAY_F + ONE_M_DECAY * esum[i];
        out_newavg[i] = na;
        out_newemb[i] = na / cs;
    }
}

// ---------------- launcher ----------------
extern "C" void kernel_launch(void* const* d_in, const int* in_sizes, int n_in,
                              void* d_out, int out_size, void* d_ws, size_t ws_size,
                              hipStream_t stream) {
    const float* z            = (const float*)d_in[0];  // (32, 64, 4096)
    const float* embedding    = (const float*)d_in[1];  // (512, 64)
    const float* cluster_size = (const float*)d_in[2];  // (512,)
    const float* embed_avg    = (const float*)d_in[3];  // (512, 64)
    float* out = (float*)d_out;

    // output tuple, flattened in return order
    const size_t OFF_Q      = 0;                          // 8388608
    const size_t OFF_IDX    = OFF_Q + (size_t)N_ROWS * EMB;
    const size_t OFF_LOSS   = OFF_IDX + N_ROWS;
    const size_t OFF_PERP   = OFF_LOSS + 1;
    const size_t OFF_NEWEMB = OFF_PERP + 1;
    const size_t OFF_NEWCS  = OFF_NEWEMB + K_CODES * EMB;
    const size_t OFF_NEWAVG = OFF_NEWCS + K_CODES;

    // workspace carving
    char* wsb = (char*)d_ws;
    int*   ws_idx  = (int*)wsb;                                            // 512 KB
    float* ws_loss = (float*)(wsb + (size_t)N_ROWS * sizeof(int));         // 512 f32
    float* ws_cnts = ws_loss + LOSS_PARTS;                                 // 512 f32
    float* ws_esum = ws_cnts + K_CODES;                                    // 32768 f32
    float* ws_e2   = ws_esum + (size_t)K_CODES * EMB;                      // 512 f32

    vq_e2_kernel<<<(K_CODES + 255) / 256, 256, 0, stream>>>(embedding, ws_e2);

    vq_assign_kernel<<<N_ROWS / 256, 256, 0, stream>>>(
        z, embedding, ws_e2, out + OFF_Q, out + OFF_IDX, ws_idx, ws_loss);

    vq_segsum_kernel<<<K_CODES, 512, 0, stream>>>(ws_idx, z, ws_cnts, ws_esum);

    vq_final_kernel<<<1, 256, 0, stream>>>(
        cluster_size, embed_avg, ws_cnts, ws_esum, ws_loss,
        out + OFF_LOSS, out + OFF_PERP, out + OFF_NEWEMB,
        out + OFF_NEWCS, out + OFF_NEWAVG);
}
